// ButterflyMLP_89240830476290
// MI455X (gfx1250) — matrix operand order repaired and checked by reading, not compiled, not run on hardware
//
#include <hip/hip_runtime.h>
#include <hip/hip_bf16.h>
#include <cmath>

typedef unsigned short u16;
typedef unsigned int   u32;

typedef __attribute__((ext_vector_type(16))) __bf16 v16bf;
typedef __attribute__((ext_vector_type(8)))  float  v8f;

// ---------- bf16 helpers (round-to-nearest-even, bit level -> no codegen risk) ----------
__device__ __forceinline__ u16 f2bf(float f) {
    u32 u = __float_as_uint(f);
    u += 0x7fffu + ((u >> 16) & 1u);
    return (u16)(u >> 16);
}
__device__ __forceinline__ float bf2f(u32 h) { return __uint_as_float(h << 16); }

// WGP-scope prefetch (SCOPE field = 0): pulls the line into ALL cache levels,
// including the WGP cache. __builtin_prefetch can only reach SE/SYS scope,
// which per the ISA skips the WGP$ — so use inline asm for the near-scope form.
__device__ __forceinline__ void prefetch_wgp(const void* p) {
    asm volatile("global_prefetch_b8 %0, off" ::"v"(p) : "memory");
}

// ---------- elementwise fp32 -> bf16 ----------
__global__ __launch_bounds__(256) void convert_f32_bf16(const float* __restrict__ in,
                                                        u16* __restrict__ out, int n) {
    int i = (blockIdx.x * 256 + threadIdx.x) * 4;
    if (i + 3 < n) {
        float4 v = *(const float4*)(in + i);
        u32 lo = (u32)f2bf(v.x) | ((u32)f2bf(v.y) << 16);
        u32 hi = (u32)f2bf(v.z) | ((u32)f2bf(v.w) << 16);
        *(uint2*)(out + i) = make_uint2(lo, hi);
    }
}

// ---------- tiled transpose fp32(rows x cols) -> bf16(cols x rows) ----------
__global__ __launch_bounds__(256) void transpose_bf16(const float* __restrict__ in,
                                                      u16* __restrict__ out,
                                                      int rows, int cols) {
    __shared__ float tile[32][33];
    int c0 = blockIdx.x * 32, r0 = blockIdx.y * 32;
    int tx = threadIdx.x, ty = threadIdx.y;  // block (32, 8)
#pragma unroll
    for (int j = 0; j < 4; ++j)
        tile[ty + j * 8][tx] = in[(size_t)(r0 + ty + j * 8) * cols + c0 + tx];
    __syncthreads();
#pragma unroll
    for (int j = 0; j < 4; ++j)
        out[(size_t)(c0 + ty + j * 8) * rows + r0 + tx] = f2bf(tile[tx][ty + j * 8]);
}

// ---------- WMMA bf16 GEMM: C(MxN) = A(MxK) * Bt(NxK)^T + bias ----------
// Block: 256 thr = 8 waves as 2(M) x 4(N). Wave tile: 64(M) x 32(N) = 4x2 WMMA tiles.
// Block tile: 128 x 128.
union FragU { v16bf v; uint4 u[2]; };

template <bool OUT_BF16>
__global__ __launch_bounds__(256) void gemm_wmma_bf16(
    const u16* __restrict__ A,   // M x K, row-major bf16
    const u16* __restrict__ Bt,  // N x K, row-major bf16 (i.e. B transposed)
    const float* __restrict__ bias0,
    const float* __restrict__ bias1,  // may be null
    void* __restrict__ Cout, int M, int N, int K) {
    const int lane  = threadIdx.x & 31;
    const int wave  = threadIdx.x >> 5;
    const int wm    = wave >> 2;  // 0..1
    const int wn    = wave & 3;   // 0..3
    const int row16 = lane & 15;
    const int half  = lane >> 4;
    const int bm = blockIdx.y * 128 + wm * 64;
    const int bn = blockIdx.x * 128 + wn * 32;

    v8f acc[4][2];
#pragma unroll
    for (int i = 0; i < 4; ++i)
#pragma unroll
        for (int j = 0; j < 2; ++j) acc[i][j] = (v8f)0.0f;

    // A 16-bit layout: lane<16 holds K{0..7,16..23}, lane>=16 holds K{8..15,24..31}
    const u16* Arow[4];
#pragma unroll
    for (int i = 0; i < 4; ++i)
        Arow[i] = A + (size_t)(bm + i * 16 + row16) * K + half * 8;
    // B 16-bit layout: lane = column, lane<16 holds K{0..7,16..23}, lane>=16 holds K{8..15,24..31}
    const u16* Brow[2];
#pragma unroll
    for (int j = 0; j < 2; ++j)
        Brow[j] = Bt + (size_t)(bn + j * 16 + row16) * K + half * 8;

    for (int k = 0; k < K; k += 32) {
        // WGP-scope prefetch of the K-chunk 4 iterations (128 elements) ahead (per-lane addresses cover the tile rows)
        prefetch_wgp(Arow[0] + k + 128);
        prefetch_wgp(Brow[0] + k + 128);
        v16bf a[4], b[2];
#pragma unroll
        for (int i = 0; i < 4; ++i) {
            const uint4* p = (const uint4*)(Arow[i] + k);
            FragU f;
            f.u[0] = p[0];      // K half-block 0
            f.u[1] = p[2];      // +16 elements
            a[i] = f.v;
        }
#pragma unroll
        for (int j = 0; j < 2; ++j) {
            const uint4* p = (const uint4*)(Brow[j] + k);
            FragU f;
            f.u[0] = p[0];
            f.u[1] = p[2];      // +16 elements
            b[j] = f.v;
        }
#pragma unroll
        for (int i = 0; i < 4; ++i)
#pragma unroll
            for (int j = 0; j < 2; ++j)
                acc[i][j] = __builtin_amdgcn_wmma_f32_16x16x32_bf16(
                    false, a[i], false, b[j], (short)0, acc[i][j], false, false);
    }

    // Epilogue: C/D layout -> VGPR r holds row (r + 8*half), col = lane&15
#pragma unroll
    for (int j = 0; j < 2; ++j) {
        const int col = bn + j * 16 + row16;
        float bsum = bias0[col];
        if (bias1) bsum += bias1[col];
#pragma unroll
        for (int i = 0; i < 4; ++i) {
#pragma unroll
            for (int r = 0; r < 8; ++r) {
                const int row = bm + i * 16 + r + half * 8;
                float v = acc[i][j][r] + bsum;
                if (OUT_BF16)
                    ((u16*)Cout)[(size_t)row * N + col] = f2bf(v);
                else
                    ((float*)Cout)[(size_t)row * N + col] = v;
            }
        }
    }
}

// ---------- butterfly-up + bias + exact GELU + butterfly-down ----------
// One block = 8 tokens, 256 threads, 8*4096 floats of dynamic LDS (128 KB).
// Each stage: thread owns 8 pairs; the 2x2 weight (one float4) is reused for 8 tokens.
#define BT_TOK 4096

__device__ __forceinline__ void butterfly12(float* __restrict__ buf,
                                            const float4* __restrict__ w, int tid) {
#pragma unroll
    for (int s = 0; s < 12; ++s) {
        __syncthreads();
        const int stride = 1 << s;
#pragma unroll
        for (int i = 0; i < 8; ++i) {
            const int p  = tid + i * 256;        // pair index 0..2047
            const int b  = p >> s;               // p / stride
            const int j  = p & (stride - 1);
            const int i0 = (b << (s + 1)) + j;
            const int i1 = i0 + stride;
            const float4 wv = w[s * 2048 + p];
#pragma unroll
            for (int t = 0; t < 8; ++t) {
                float* bt = buf + t * BT_TOK;
                const float av = bt[i0], cv = bt[i1];
                bt[i0] = fmaf(wv.x, av, wv.y * cv);
                bt[i1] = fmaf(wv.z, av, wv.w * cv);
            }
        }
    }
}

__global__ __launch_bounds__(256) void butterfly_gelu(
    const u16* __restrict__ Hin,          // 8192 x 4096 bf16
    const float* __restrict__ upw,        // 12 x 2048 x 2 x 2
    const float* __restrict__ upb,        // 4096
    const float* __restrict__ dnw,        // 12 x 2048 x 2 x 2
    u16* __restrict__ Hout) {             // 8192 x 4096 bf16
    extern __shared__ float buf[];        // 8 * 4096 floats
    const int tid = threadIdx.x;
    const size_t tok0 = (size_t)blockIdx.x * 8;

    // load + expand bf16 -> f32
#pragma unroll
    for (int t = 0; t < 8; ++t) {
        const uint4* src = (const uint4*)(Hin + (tok0 + t) * 4096);
        float* bt = buf + t * BT_TOK;
#pragma unroll
        for (int i = 0; i < 2; ++i) {
            const int q = tid + i * 256;
            uint4 v = src[q];
            const int e = q * 8;
            u32 w0 = v.x, w1 = v.y, w2 = v.z, w3 = v.w;
            bt[e + 0] = bf2f(w0 & 0xffffu); bt[e + 1] = bf2f(w0 >> 16);
            bt[e + 2] = bf2f(w1 & 0xffffu); bt[e + 3] = bf2f(w1 >> 16);
            bt[e + 4] = bf2f(w2 & 0xffffu); bt[e + 5] = bf2f(w2 >> 16);
            bt[e + 6] = bf2f(w3 & 0xffffu); bt[e + 7] = bf2f(w3 >> 16);
        }
    }

    butterfly12(buf, (const float4*)upw, tid);

    __syncthreads();
#pragma unroll
    for (int i = 0; i < 16; ++i) {
        const int e = tid + i * 256;
        const float bb = upb[e];
#pragma unroll
        for (int t = 0; t < 8; ++t) {
            float v = buf[t * BT_TOK + e] + bb;
            buf[t * BT_TOK + e] = 0.5f * v * (1.0f + erff(v * 0.70710678118654752f));
        }
    }

    butterfly12(buf, (const float4*)dnw, tid);

    __syncthreads();
    // pack f32 -> bf16 and store
#pragma unroll
    for (int t = 0; t < 8; ++t) {
        uint4* dst = (uint4*)(Hout + (tok0 + t) * 4096);
        const float* bt = buf + t * BT_TOK;
#pragma unroll
        for (int i = 0; i < 2; ++i) {
            const int q = tid + i * 256;
            const int e = q * 8;
            uint4 v;
            v.x = (u32)f2bf(bt[e + 0]) | ((u32)f2bf(bt[e + 1]) << 16);
            v.y = (u32)f2bf(bt[e + 2]) | ((u32)f2bf(bt[e + 3]) << 16);
            v.z = (u32)f2bf(bt[e + 4]) | ((u32)f2bf(bt[e + 5]) << 16);
            v.w = (u32)f2bf(bt[e + 6]) | ((u32)f2bf(bt[e + 7]) << 16);
            dst[q] = v;
        }
    }
}

// ---------- host side ----------
extern "C" void kernel_launch(void* const* d_in, const int* in_sizes, int n_in,
                              void* d_out, int out_size, void* d_ws, size_t ws_size,
                              hipStream_t stream) {
    (void)in_sizes; (void)n_in; (void)out_size; (void)ws_size;
    const float* x     = (const float*)d_in[0];  // 8192 x 1024
    const float* W_exp = (const float*)d_in[1];  // 1024 x 4096
    const float* b_exp = (const float*)d_in[2];  // 4096
    const float* up_w  = (const float*)d_in[3];  // 12 x 2048 x 2 x 2
    const float* up_b  = (const float*)d_in[4];  // 4096
    const float* dn_w  = (const float*)d_in[5];  // 12 x 2048 x 2 x 2
    const float* W_con = (const float*)d_in[6];  // 4096 x 1024
    const float* b_con = (const float*)d_in[7];  // 1024
    const float* dn_b  = (const float*)d_in[8];  // 1024

    const int M = 8192, D = 1024, H = 4096;

    char* ws = (char*)d_ws;
    u16* xbf   = (u16*)ws; ws += (size_t)M * D * 2;
    u16* wexpT = (u16*)ws; ws += (size_t)H * D * 2;   // H x D (K-contiguous)
    u16* wconT = (u16*)ws; ws += (size_t)D * H * 2;   // D x H (K-contiguous)
    u16* h1    = (u16*)ws; ws += (size_t)M * H * 2;
    u16* h2    = (u16*)ws;

    // 1) x -> bf16
    convert_f32_bf16<<<(M * D) / (256 * 4), 256, 0, stream>>>(x, xbf, M * D);
    // 2) W_exp (D x H) -> wexpT (H x D) bf16 ; W_con (H x D) -> wconT (D x H) bf16
    transpose_bf16<<<dim3(H / 32, D / 32), dim3(32, 8), 0, stream>>>(W_exp, wexpT, D, H);
    transpose_bf16<<<dim3(D / 32, H / 32), dim3(32, 8), 0, stream>>>(W_con, wconT, H, D);
    // 3) h1 = x @ W_exp + b_exp   (bf16 out)
    gemm_wmma_bf16<true><<<dim3(H / 128, M / 128), 256, 0, stream>>>(
        xbf, wexpT, b_exp, nullptr, h1, M, H, D);
    // 4) h2 = butterfly_down(gelu(butterfly_up(h1) + up_bias))
    butterfly_gelu<<<M / 8, 256, 8 * BT_TOK * sizeof(float), stream>>>(
        h1, up_w, up_b, dn_w, h2);
    // 5) y = h2 @ W_con + b_con + down_bias   (fp32 out)
    gemm_wmma_bf16<false><<<dim3(D / 128, M / 128), 256, 0, stream>>>(
        h2, wconT, b_con, dn_b, (float*)d_out, M, D, H);
}
